// LSTM_67027259622087
// MI455X (gfx1250) — compile-verified, matrix operand order checked by reading, not run on hardware
//
#include <hip/hip_runtime.h>

// ---------------------------------------------------------------------------
// LSTM (B=128, T=1024, I=64, H=512, O=1) persistent-kernel for gfx1250.
//
// Design (see analysis): 64 persistent workgroups = 4 batch tiles (M=32) x
// 16 hidden chunks (32 h-cols -> 128 gate rows each). W_hh/W_ih slices are
// converted to f16 and pre-packed in LDS in native WMMA B-fragment layout.
// Each timestep: stage h-tile (f16, double-buffered in d_ws) + x-tile into
// WMMA A-fragment layout in LDS, run v_wmma_f32_16x16x32_f16 (36 per wave),
// apply gate activations, keep c in registers, write h (f16) for the next
// step, fuse the final FC via wave shuffle-reduce + atomicAdd, then cross a
// device-scope sense-reversing barrier. 1024 iterations, one kernel launch.
// ---------------------------------------------------------------------------

typedef __attribute__((ext_vector_type(16))) _Float16 v16h;
typedef __attribute__((ext_vector_type(8)))  _Float16 v8h;
typedef __attribute__((ext_vector_type(8)))  float    v8f;

#define B_       128
#define T_       1024
#define I_       64
#define H_       512
#define NWG      64          // 4 batch tiles x 16 hidden chunks
#define SMEM_BYTES 201344    // whh 128K + wih 16K + hF 32K + xF 4K + gates 16K + bias 512 + wfc 128

// LDS carve offsets (bytes)
#define OFF_WHH   0
#define OFF_WIH   131072
#define OFF_HF    147456
#define OFF_XF    180224
#define OFF_GATES 184320
#define OFF_BIAS  200704
#define OFF_WFC   201216

__device__ inline v8f wmma_f16(v16h a, v16h b, v8f c) {
  // emits v_wmma_f32_16x16x32_f16
  return __builtin_amdgcn_wmma_f32_16x16x32_f16(false, a, false, b,
                                                (short)0, c, false, false);
}

__device__ inline float fast_sigmoid(float v) {
  v = fminf(fmaxf(v, -30.f), 30.f);
  return 1.f / (1.f + __expf(-v));
}
__device__ inline float fast_tanh(float v) {
  v = fminf(fmaxf(v, -15.f), 15.f);
  float e = __expf(2.f * v);
  return (e - 1.f) / (e + 1.f);
}

// read 8 consecutive f32 from global, convert to f16, store 16B into LDS
__device__ inline void cvt_store8(const float* __restrict__ src, _Float16* dst) {
  const float4 a = *(const float4*)(src);
  const float4 b = *(const float4*)(src + 4);
  v8h v;
  v[0] = (_Float16)a.x; v[1] = (_Float16)a.y; v[2] = (_Float16)a.z; v[3] = (_Float16)a.w;
  v[4] = (_Float16)b.x; v[5] = (_Float16)b.y; v[6] = (_Float16)b.z; v[7] = (_Float16)b.w;
  *(v8h*)dst = v;
}
// raw 16B copy (8 f16) global -> LDS
__device__ inline void copy16B(const _Float16* __restrict__ src, _Float16* dst) {
  *(uint4*)dst = *(const uint4*)src;
}

// device-wide sense(generation) barrier; bar[0]=arrive count, bar[1]=generation
__device__ inline void grid_barrier(unsigned* bar, unsigned target) {
  __threadfence();
  __syncthreads();
  if (threadIdx.x == 0) {
    unsigned prev = __hip_atomic_fetch_add(&bar[0], 1u, __ATOMIC_ACQ_REL,
                                           __HIP_MEMORY_SCOPE_AGENT);
    if (prev == NWG - 1) {
      __hip_atomic_store(&bar[0], 0u, __ATOMIC_RELAXED, __HIP_MEMORY_SCOPE_AGENT);
      __hip_atomic_fetch_add(&bar[1], 1u, __ATOMIC_RELEASE, __HIP_MEMORY_SCOPE_AGENT);
    } else {
      while (__hip_atomic_load(&bar[1], __ATOMIC_ACQUIRE,
                               __HIP_MEMORY_SCOPE_AGENT) < target) {
        __builtin_amdgcn_s_sleep(2);
      }
    }
  }
  __syncthreads();
  __threadfence();
}

__global__ void lstm_init(float* __restrict__ y, const float* __restrict__ b_fc,
                          _Float16* __restrict__ hbuf, unsigned* __restrict__ bar) {
  int idx = blockIdx.x * blockDim.x + threadIdx.x;
  if (idx == 0) { bar[0] = 0u; bar[1] = 0u; }
  if (idx < B_ * T_) {              // 131072: covers y and both h buffers
    y[idx] = b_fc[0];
    hbuf[idx] = (_Float16)0.0f;     // 2 * 128*512 = 131072 f16
  }
}

__global__ __launch_bounds__(256, 1) void lstm_persistent(
    const float* __restrict__ x,     // [B,T,I]
    const float* __restrict__ W_ih,  // [4H,I]
    const float* __restrict__ W_hh,  // [4H,H]
    const float* __restrict__ b_ih,  // [4H]
    const float* __restrict__ b_hh,  // [4H]
    const float* __restrict__ W_fc,  // [1,H]
    _Float16* __restrict__ hbuf,     // [2][B][H] f16 (workspace)
    float* __restrict__ y,           // [B,T] f32 (pre-set to b_fc)
    unsigned* __restrict__ bar) {
  extern __shared__ __align__(16) char smem[];
  _Float16* whhF  = (_Float16*)(smem + OFF_WHH);   // 8 ntile x 16 kstep frags
  _Float16* wihF  = (_Float16*)(smem + OFF_WIH);   // 8 ntile x 2 kstep frags
  _Float16* hF    = (_Float16*)(smem + OFF_HF);    // 2 mtile x 16 kstep frags
  _Float16* xF    = (_Float16*)(smem + OFF_XF);    // 2 mtile x 2 kstep frags
  float*    gatesL = (float*)(smem + OFF_GATES);   // [32][128]
  float*    biasL  = (float*)(smem + OFF_BIAS);    // [128]
  float*    wfcL   = (float*)(smem + OFF_WFC);     // [32]

  const int tid  = threadIdx.x;
  const int lane = tid & 31;
  const int w    = tid >> 5;          // wave id == WMMA N-tile (0..7)
  const int mb   = blockIdx.x >> 4;   // batch tile (0..3), 32 rows each
  const int hc   = blockIdx.x & 15;   // hidden chunk (0..15), 32 cols each

  // ---- one-time: pack W_hh / W_ih slices into WMMA B-fragment layout ------
  // Fragment lane L holds N = (L&15); elements p: K = p + (p&8) + 8*(L>>4).
  // Each lane's 16 f16 = two 8-element runs at k0 and k0+16.
#pragma unroll 4
  for (int i = 0; i < 32; ++i) {      // 8192 runs of 8 f32 -> 8 f16
    int R = tid + (i << 8);
    int u = R & 1, L = (R >> 1) & 31, kb = (R >> 6) & 15, nt = R >> 10;
    int n  = (nt << 4) + (L & 15);                      // local gate col 0..127
    int g  = (n >> 5) * H_ + hc * 32 + (n & 31);        // global gate row
    int k0 = (kb << 5) + (u << 4) + ((L >> 4) << 3);
    cvt_store8(W_hh + (size_t)g * H_ + k0,
               whhF + ((((nt << 4) + kb) << 5) + L) * 16 + (u << 3));
  }
#pragma unroll
  for (int i = 0; i < 4; ++i) {       // 1024 runs
    int R = tid + (i << 8);
    int u = R & 1, L = (R >> 1) & 31, kb = (R >> 6) & 1, nt = R >> 7;
    int n  = (nt << 4) + (L & 15);
    int g  = (n >> 5) * H_ + hc * 32 + (n & 31);
    int k0 = (kb << 5) + (u << 4) + ((L >> 4) << 3);
    cvt_store8(W_ih + (size_t)g * I_ + k0,
               wihF + ((((nt << 1) + kb) << 5) + L) * 16 + (u << 3));
  }
  if (tid < 128) {
    int g = (tid >> 5) * H_ + hc * 32 + (tid & 31);
    biasL[tid] = b_ih[g] + b_hh[g];
  }
  if (tid < 32) wfcL[tid] = W_fc[hc * 32 + tid];

  float creg[4] = {0.f, 0.f, 0.f, 0.f};   // cell state, registers only

  for (int t = 0; t < T_; ++t) {
    const _Float16* hsrc = hbuf + ((t & 1) * (B_ * H_)) + mb * 32 * H_;
    _Float16* hdst       = hbuf + (((t + 1) & 1) * (B_ * H_)) + mb * 32 * H_;

    // ---- stage h tile (32x512 f16) into A-fragment layout ----------------
#pragma unroll
    for (int i = 0; i < 8; ++i) {     // 2048 x 16B
      int R = tid + (i << 8);
      int u = R & 1, L = (R >> 1) & 31, kb = (R >> 6) & 15, mt = R >> 10;
      int row = (mt << 4) + (L & 15);
      int col = (kb << 5) + (u << 4) + ((L >> 4) << 3);
      copy16B(hsrc + (row << 9) + col,
              hF + ((((mt << 4) + kb) << 5) + L) * 16 + (u << 3));
    }
    // ---- stage x tile (32x64 f32 -> f16) ---------------------------------
    {
      int R = tid;                    // 256 runs
      int u = R & 1, L = (R >> 1) & 31, kb = (R >> 6) & 1, mt = R >> 7;
      int row = (mt << 4) + (L & 15);
      int col = (kb << 5) + (u << 4) + ((L >> 4) << 3);
      cvt_store8(x + (size_t)(mb * 32 + row) * (T_ * I_) + t * I_ + col,
                 xF + ((((mt << 1) + kb) << 5) + L) * 16 + (u << 3));
    }
    __syncthreads();

    // ---- gates[32 x 16] tile per wave: h@Whh^T + x@Wih^T -----------------
    v8f acc0 = {0.f, 0.f, 0.f, 0.f, 0.f, 0.f, 0.f, 0.f};
    v8f acc1 = {0.f, 0.f, 0.f, 0.f, 0.f, 0.f, 0.f, 0.f};
#pragma unroll
    for (int kb = 0; kb < 16; ++kb) {
      v16h bf = *(const v16h*)(whhF + ((((w << 4) + kb) << 5) + lane) * 16);
      v16h a0 = *(const v16h*)(hF + (((kb) << 5) + lane) * 16);
      v16h a1 = *(const v16h*)(hF + (((16 + kb) << 5) + lane) * 16);
      acc0 = wmma_f16(a0, bf, acc0);
      acc1 = wmma_f16(a1, bf, acc1);
    }
#pragma unroll
    for (int kb = 0; kb < 2; ++kb) {
      v16h bf = *(const v16h*)(wihF + ((((w << 1) + kb) << 5) + lane) * 16);
      v16h a0 = *(const v16h*)(xF + ((kb << 5) + lane) * 16);
      v16h a1 = *(const v16h*)(xF + (((2 + kb) << 5) + lane) * 16);
      acc0 = wmma_f16(a0, bf, acc0);
      acc1 = wmma_f16(a1, bf, acc1);
    }
    // scatter C-frags (+bias) to LDS: D elem r -> M = 8*(lane>>4)+r, N = lane&15
    {
      int hlf = lane >> 4;
      int n   = (w << 4) + (lane & 15);
      float bs = biasL[n];
#pragma unroll
      for (int r = 0; r < 8; ++r) {
        gatesL[(((hlf << 3) + r)) * 128 + n]      = acc0[r] + bs;
        gatesL[((16 + (hlf << 3) + r)) * 128 + n] = acc1[r] + bs;
      }
    }
    __syncthreads();

    // ---- elementwise LSTM cell + h store + fused FC ----------------------
    {
      int jloc = lane;                 // hidden col within chunk (0..31)
      float wfc = wfcL[jloc];
#pragma unroll
      for (int r = 0; r < 4; ++r) {
        int b = w + (r << 3);          // batch row within tile (0..31)
        float gi = gatesL[b * 128 +       jloc];
        float gf = gatesL[b * 128 + 32  + jloc];
        float gg = gatesL[b * 128 + 64  + jloc];
        float go = gatesL[b * 128 + 96  + jloc];
        float iv = fast_sigmoid(gi);
        float fv = fast_sigmoid(gf);
        float gv = fast_tanh(gg);
        float ov = fast_sigmoid(go);
        float cn = fv * creg[r] + iv * gv;
        creg[r] = cn;
        float hv = ov * fast_tanh(cn);
        hdst[(b << 9) + hc * 32 + jloc] = (_Float16)hv;
        float part = hv * wfc;         // fused y += h . W_fc (chunk partial)
#pragma unroll
        for (int m = 16; m > 0; m >>= 1) part += __shfl_xor(part, m, 32);
        if (jloc == 0)
          atomicAdd(y + (size_t)(mb * 32 + b) * T_ + t, part);
      }
    }

    grid_barrier(bar, (unsigned)(t + 1));
  }
}

extern "C" void kernel_launch(void* const* d_in, const int* in_sizes, int n_in,
                              void* d_out, int out_size, void* d_ws, size_t ws_size,
                              hipStream_t stream) {
  (void)in_sizes; (void)n_in; (void)out_size; (void)ws_size;
  const float* x    = (const float*)d_in[0];
  const float* W_ih = (const float*)d_in[1];
  const float* W_hh = (const float*)d_in[2];
  const float* b_ih = (const float*)d_in[3];
  const float* b_hh = (const float*)d_in[4];
  const float* W_fc = (const float*)d_in[5];
  const float* b_fc = (const float*)d_in[6];
  float* y = (float*)d_out;

  unsigned* bar   = (unsigned*)d_ws;                       // 2 x u32
  _Float16* hbuf  = (_Float16*)((char*)d_ws + 256);        // 2*128*512 f16

  lstm_init<<<512, 256, 0, stream>>>(y, b_fc, hbuf, bar);

  // opt into >64KB dynamic LDS (gfx1250 allows up to 320KB/WG); attribute
  // setters are legal during graph capture, and the call is idempotent.
  (void)hipFuncSetAttribute(reinterpret_cast<const void*>(lstm_persistent),
                            hipFuncAttributeMaxDynamicSharedMemorySize,
                            SMEM_BYTES);

  lstm_persistent<<<NWG, 256, SMEM_BYTES, stream>>>(
      x, W_ih, W_hh, b_ih, b_hh, W_fc, hbuf, y, bar);
}
